// ClusterMemory_50148038148624
// MI455X (gfx1250) — compile-verified, matrix operand order checked by reading, not run on hardware
//
#include <hip/hip_runtime.h>
#include <stdint.h>

// ---- problem constants ----
#define B_       256
#define D_       2048
#define N_       8192
#define P_       8
#define G_       9          // 1 main + 8 parts
#define INV_TEMP 20.0f      // 1/0.05
#define MOM      0.2f
#define NSTRIP   128        // N / 64
#define KCH      64         // D / 32 (K chunks of 32 for bf16 WMMA)
#define LSTRIDE  20         // LDS row stride in dwords (bf16 pairs), 16B-aligned, conflict-free

typedef __attribute__((ext_vector_type(16))) __bf16 v16bf;
typedef __attribute__((ext_vector_type(8)))  float  v8f;
typedef __attribute__((ext_vector_type(2)))  __bf16 v2bf;

union V8 { v8f v; float f[8]; };
union AB { uint4 q[2]; uint32_t u[8]; v16bf v; };

// f32 pair -> packed bf16 dword. Prefer HW packed convert; else round-half-up
// (+0x8000) and splice the two high halves with a single v_perm_b32.
__device__ __forceinline__ uint32_t pk_bf16(float lo, float hi) {
#if __has_builtin(__builtin_amdgcn_cvt_pk_bf16_f32)
  union { v2bf v; uint32_t u; } c;
  c.v = __builtin_amdgcn_cvt_pk_bf16_f32(lo, hi);
  return c.u;
#else
  const uint32_t a = __float_as_uint(lo) + 0x8000u;
  const uint32_t b = __float_as_uint(hi) + 0x8000u;
  // pool: bytes 0-3 = src1 (a), bytes 4-7 = src0 (b); take hi16 of each
  return __builtin_amdgcn_perm(b, a, 0x07060302u);
#endif
}

__device__ __forceinline__ uint4 pk_bf16x8(float4 f0, float4 f1) {
  uint4 r;
  r.x = pk_bf16(f0.x, f0.y); r.y = pk_bf16(f0.z, f0.w);
  r.z = pk_bf16(f1.x, f1.y); r.w = pk_bf16(f1.z, f1.w);
  return r;
}

// ---------------------------------------------------------------------------
// Kernel 1: L2-normalize x rows; emit fp32 x_norm and bf16 x packed in the
// exact CDNA5 WMMA A-matrix (16x32, 16-bit) per-lane layout:
//   lanes 0-15 : K0..7 (v0..3), K16..23 (v4..7);  lanes 16-31 : K8..15, K24..31
// ---------------------------------------------------------------------------
__global__ __launch_bounds__(256)
void cm_norm_pack(const float* __restrict__ inputs,
                  float* __restrict__ xnorm, uint32_t* __restrict__ xpack) {
  __shared__ float red[256];
  const int b = blockIdx.x, t = threadIdx.x;
  const float4* src = reinterpret_cast<const float4*>(inputs + (size_t)b * D_ + t * 8);
  float4 f0 = src[0], f1 = src[1];
  float v[8] = {f0.x, f0.y, f0.z, f0.w, f1.x, f1.y, f1.z, f1.w};
  float ss = 0.f;
#pragma unroll
  for (int i = 0; i < 8; i++) ss += v[i] * v[i];
  red[t] = ss; __syncthreads();
  for (int o = 128; o > 0; o >>= 1) { if (t < o) red[t] += red[t + o]; __syncthreads(); }
  const float inv = 1.0f / fmaxf(sqrtf(red[0]), 1e-12f);
#pragma unroll
  for (int i = 0; i < 8; i++) v[i] *= inv;

  float4* dx = reinterpret_cast<float4*>(xnorm + (size_t)b * D_ + t * 8);
  dx[0] = make_float4(v[0], v[1], v[2], v[3]);
  dx[1] = make_float4(v[4], v[5], v[6], v[7]);

  // pack 8 contiguous K into 4 dwords of the A layout
  const int k0 = t * 8;
  const int kc = k0 >> 5;
  const int group = (k0 & 31) >> 3;               // which K-octet in chunk
  const int lane = (b & 15) + ((group & 1) << 4);
  const int regb = (group >> 1) * 4;
  uint4 pk;
  pk.x = pk_bf16(v[0], v[1]); pk.y = pk_bf16(v[2], v[3]);
  pk.z = pk_bf16(v[4], v[5]); pk.w = pk_bf16(v[6], v[7]);
  *reinterpret_cast<uint4*>(
      xpack + ((((size_t)(b >> 4) * KCH + kc) * 32 + lane) * 8 + regb)) = pk;
}

// ---------------------------------------------------------------------------
// Kernel 2: GEMM + online-softmax partials. One WG per (matrix g, 64-wide N
// strip); 8 waves, wave w owns M rows [w*32, w*32+32) -> 2x4 tiles of 16x16.
// Feat tile is converted to bf16 ONCE at LDS-store time (packed pair dwords),
// so the consume path is pure ds_load_b128 -> v_wmma. Double-buffered LDS
// (pointer-swapped, K loop forced unroll(1) so accumulators stay in place),
// one barrier per K chunk. Each part_features element read once from HBM.
// ---------------------------------------------------------------------------
__global__ __launch_bounds__(256)
void cm_gemm_lse(const uint32_t* __restrict__ xpack,
                 const float* __restrict__ features,
                 const float* __restrict__ part_features,
                 float* __restrict__ pmax, float* __restrict__ psum) {
  __shared__ uint32_t lb[2][64 * LSTRIDE];        // bf16-pair tiles, double buffer
  const int g = blockIdx.y;
  const int strip = blockIdx.x;
  const int n0 = strip * 64;
  const float* feat = (g == 0) ? features
                               : part_features + (size_t)(g - 1) * N_ * D_;
  const int tid = threadIdx.x;
  const int w = tid >> 5, l = tid & 31, lh = l & 15, half = l >> 4;

  // loader mapping: 4 threads per row, 8 contiguous floats each
  const int lrow = tid >> 2, lc8 = tid & 3;
  const float* gsrc = feat + (size_t)(n0 + lrow) * D_ + lc8 * 8;

  // double-buffer pointers (swapped each iteration; no unroll -> no acc copies)
  uint32_t* st_cur = &lb[0][lrow * LSTRIDE + lc8 * 4];
  uint32_t* st_nxt = &lb[1][lrow * LSTRIDE + lc8 * 4];
  const uint32_t* ld_cur = &lb[0][(lh)*LSTRIDE + half * 4];
  const uint32_t* ld_nxt = &lb[1][(lh)*LSTRIDE + half * 4];

  V8 acc[2][4];
#pragma unroll
  for (int t = 0; t < 2; t++)
#pragma unroll
    for (int nt = 0; nt < 4; nt++)
#pragma unroll
      for (int r = 0; r < 8; r++) acc[t][nt].f[r] = 0.0f;

  // prologue: stage K-chunk 0
  {
    const float4 f0 = *reinterpret_cast<const float4*>(gsrc);
    const float4 f1 = *reinterpret_cast<const float4*>(gsrc + 4);
    *reinterpret_cast<uint4*>(st_cur) = pk_bf16x8(f0, f1);
  }
  __syncthreads();

#pragma unroll 1
  for (int kc = 0; kc < KCH; kc++) {
    // issue next chunk's global loads early (latency hidden behind WMMAs)
    float4 g0, g1;
    const bool more = (kc + 1 < KCH);
    if (more) {
      const float* gp = gsrc + (kc + 1) * 32;
      g0 = *reinterpret_cast<const float4*>(gp);
      g1 = *reinterpret_cast<const float4*>(gp + 4);
    }

    // A tiles: pre-packed bf16 register image, L2-resident
    AB a[2];
#pragma unroll
    for (int t = 0; t < 2; t++) {
      const int mb = w * 2 + t;
      const uint4* ap = reinterpret_cast<const uint4*>(
          xpack + (((size_t)mb * KCH + kc) * 32 + l) * 8);
      a[t].q[0] = ap[0]; a[t].q[1] = ap[1];
    }

    // B tiles: preload all four register images (8x ds_load_b128, one clause)
    AB bm[4];
#pragma unroll
    for (int nt = 0; nt < 4; nt++) {
      const uint32_t* bp = ld_cur + nt * 16 * LSTRIDE;
      bm[nt].q[0] = *reinterpret_cast<const uint4*>(bp);      // K = half*8 .. +7
      bm[nt].q[1] = *reinterpret_cast<const uint4*>(bp + 8);  // K = 16+half*8 .. +7
    }
#pragma unroll
    for (int nt = 0; nt < 4; nt++)
#pragma unroll
      for (int t = 0; t < 2; t++)
        acc[t][nt].v = __builtin_amdgcn_wmma_f32_16x16x32_bf16(
            false, a[t].v, false, bm[nt].v, (short)0, acc[t][nt].v, false, false);

    // stage next chunk into the other buffer, then swap buffers
    if (more)
      *reinterpret_cast<uint4*>(st_nxt) = pk_bf16x8(g0, g1);
    uint32_t* ts = st_cur; st_cur = st_nxt; st_nxt = ts;
    const uint32_t* tl = ld_cur; ld_cur = ld_nxt; ld_nxt = tl;
    __syncthreads();
  }

  // online softmax partials: per row, max & sum(exp) over this 64-col strip.
  // C layout: reg r -> M=r (lanes 0-15) / M=8+r (lanes 16-31); N = lane%16.
#pragma unroll
  for (int t = 0; t < 2; t++) {
#pragma unroll
    for (int r = 0; r < 8; r++) {
      float vv[4]; float mx = -3.4e38f;
#pragma unroll
      for (int nt = 0; nt < 4; nt++) {
        vv[nt] = acc[t][nt].f[r] * INV_TEMP;
        mx = fmaxf(mx, vv[nt]);
      }
#pragma unroll
      for (int off = 1; off < 16; off <<= 1)
        mx = fmaxf(mx, __shfl_xor(mx, off, 32));    // stays within 16-lane half
      float sm = 0.f;
#pragma unroll
      for (int nt = 0; nt < 4; nt++) sm += __expf(vv[nt] - mx);
#pragma unroll
      for (int off = 1; off < 16; off <<= 1) sm += __shfl_xor(sm, off, 32);
      if (lh == 0) {
        const int m = w * 32 + t * 16 + half * 8 + r;
        const size_t idx = ((size_t)g * NSTRIP + strip) * B_ + m;
        pmax[idx] = mx; psum[idx] = sm;
      }
    }
  }
}

// ---------------------------------------------------------------------------
// Kernel 3: merge strip partials -> lse[g][b]
// ---------------------------------------------------------------------------
__global__ __launch_bounds__(256)
void cm_lse_reduce(const float* __restrict__ pmax, const float* __restrict__ psum,
                   float* __restrict__ lse) {
  const int g = blockIdx.x, b = threadIdx.x;
  float M = -3.4e38f;
  for (int s = 0; s < NSTRIP; s++)
    M = fmaxf(M, pmax[((size_t)g * NSTRIP + s) * B_ + b]);
  float S = 0.f;
  for (int s = 0; s < NSTRIP; s++) {
    const size_t i = ((size_t)g * NSTRIP + s) * B_ + b;
    S += psum[i] * __expf(pmax[i] - M);
  }
  lse[g * B_ + b] = M + __logf(S);
}

// ---------------------------------------------------------------------------
// Kernel 4: target-row dots + per-sample loss contribution (no atomics)
// ---------------------------------------------------------------------------
__global__ __launch_bounds__(256)
void cm_loss(const float* __restrict__ xnorm, const float* __restrict__ features,
             const float* __restrict__ part_features, const float* __restrict__ score,
             const int* __restrict__ targets, const float* __restrict__ lse,
             float* __restrict__ contrib) {
  __shared__ float red[256];
  const int b = blockIdx.x, t = threadIdx.x;
  const int tgt = targets[b];
  float c = 0.f;
  for (int g = 0; g < G_; g++) {
    const float* frow = ((g == 0) ? features
                                  : part_features + (size_t)(g - 1) * N_ * D_) +
                        (size_t)tgt * D_;
    float s = 0.f;
#pragma unroll
    for (int j = 0; j < 8; j++) {
      const int d = t + j * 256;
      s += xnorm[(size_t)b * D_ + d] * frow[d];
    }
    red[t] = s; __syncthreads();
    for (int o = 128; o > 0; o >>= 1) { if (t < o) red[t] += red[t + o]; __syncthreads(); }
    if (t == 0) {
      const float dot = red[0] * INV_TEMP;
      const float ls = lse[g * B_ + b];
      if (g == 0) c += (ls - dot) * (1.0f / B_);                       // main CE
      else        c += score[b * P_ + (g - 1)] * (ls - dot) * (1.0f / (P_ * B_));
    }
    __syncthreads();
  }
  if (t == 0) contrib[b] = c;
}

__global__ __launch_bounds__(256)
void cm_sum(const float* __restrict__ contrib, float* __restrict__ out) {
  __shared__ float red[256];
  const int t = threadIdx.x;
  red[t] = contrib[t]; __syncthreads();
  for (int o = 128; o > 0; o >>= 1) { if (t < o) red[t] += red[t + o]; __syncthreads(); }
  if (t == 0) out[0] = red[0];
}

// ---------------------------------------------------------------------------
// Kernel 5: per-cluster mean + momentum + renorm (one WG per cluster row)
// ---------------------------------------------------------------------------
__global__ __launch_bounds__(256)
void cm_update(const float* __restrict__ xnorm, const float* __restrict__ features,
               const int* __restrict__ targets, float* __restrict__ out) {
  __shared__ int list[B_]; __shared__ int cnt_s; __shared__ float red[256];
  const int n = blockIdx.x, t = threadIdx.x;
  if (t == 0) {
    int c = 0;
    for (int b = 0; b < B_; b++) if (targets[b] == n) list[c++] = b;
    cnt_s = c;
  }
  __syncthreads();
  const int cnt = cnt_s;
  const float* frow = features + (size_t)n * D_;
  float* orow = out + (size_t)n * D_;
  if (cnt == 0) {
#pragma unroll
    for (int j = 0; j < 8; j++) { const int d = t + j * 256; orow[d] = frow[d]; }
    return;
  }
  float u[8]; float ss = 0.f; const float invc = 1.0f / (float)cnt;
#pragma unroll
  for (int j = 0; j < 8; j++) {
    const int d = t + j * 256;
    float s = 0.f;
    for (int i = 0; i < cnt; i++) s += xnorm[(size_t)list[i] * D_ + d];
    const float uu = MOM * frow[d] + (1.0f - MOM) * s * invc;
    u[j] = uu; ss += uu * uu;
  }
  red[t] = ss; __syncthreads();
  for (int o = 128; o > 0; o >>= 1) { if (t < o) red[t] += red[t + o]; __syncthreads(); }
  const float inv = 1.0f / fmaxf(sqrtf(red[0]), 1e-12f);
#pragma unroll
  for (int j = 0; j < 8; j++) { const int d = t + j * 256; orow[d] = u[j] * inv; }
}

// ---------------------------------------------------------------------------
extern "C" void kernel_launch(void* const* d_in, const int* in_sizes, int n_in,
                              void* d_out, int out_size, void* d_ws, size_t ws_size,
                              hipStream_t stream) {
  (void)in_sizes; (void)n_in; (void)out_size; (void)ws_size;
  const float* inputs        = (const float*)d_in[0];
  /* d_in[1] = ema_inputs: unused by reference */
  const float* score         = (const float*)d_in[2];
  const float* part_features = (const float*)d_in[3];
  const float* features      = (const float*)d_in[4];
  const int*   targets       = (const int*)d_in[5];
  float* out = (float*)d_out;   // out[0] = loss, out[1..] = new_features [N,D]

  float*    xnorm   = (float*)d_ws;                                   // B*D f32
  uint32_t* xpack   = (uint32_t*)(xnorm + (size_t)B_ * D_);           // B*D/2 dwords
  float*    pmax    = (float*)(xpack + (size_t)B_ * D_ / 2);          // G*NSTRIP*B
  float*    psum    = pmax + (size_t)G_ * NSTRIP * B_;                // G*NSTRIP*B
  float*    lse     = psum + (size_t)G_ * NSTRIP * B_;                // G*B
  float*    contrib = lse + G_ * B_;                                  // B

  cm_norm_pack<<<dim3(B_), dim3(256), 0, stream>>>(inputs, xnorm, xpack);
  cm_gemm_lse<<<dim3(NSTRIP, G_), dim3(256), 0, stream>>>(
      xpack, features, part_features, pmax, psum);
  cm_lse_reduce<<<dim3(G_), dim3(256), 0, stream>>>(pmax, psum, lse);
  cm_loss<<<dim3(B_), dim3(256), 0, stream>>>(
      xnorm, features, part_features, score, targets, lse, contrib);
  cm_sum<<<dim3(1), dim3(256), 0, stream>>>(contrib, out);
  cm_update<<<dim3(N_), dim3(256), 0, stream>>>(xnorm, features, targets, out + 1);
}
